// _DenseAsppBlock_14345190768894
// MI455X (gfx1250) — matrix operand order repaired and checked by reading, not compiled, not run on hardware
//
#include <hip/hip_runtime.h>
#include <math.h>

// ---------------- CDNA5 WMMA types ----------------
typedef __attribute__((ext_vector_type(16))) __bf16 v16bf;
typedef __attribute__((ext_vector_type(16))) unsigned short v16u;
typedef __attribute__((ext_vector_type(8)))  unsigned short v8u;
typedef __attribute__((ext_vector_type(8)))  float v8f;

__device__ __forceinline__ unsigned short f2bf(float f) {
  unsigned u = __builtin_bit_cast(unsigned, f);
  u += 0x7FFFu + ((u >> 16) & 1u);   // round-to-nearest-even
  return (unsigned short)(u >> 16);
}
__device__ __forceinline__ unsigned pack2bf(float a, float b) {
  return (unsigned)f2bf(a) | ((unsigned)f2bf(b) << 16);
}
__device__ __forceinline__ float sigmoidf_(float x) { return 1.0f / (1.0f + expf(-x)); }

// combine two 16-byte LDS chunks into one 16-element fragment
__device__ __forceinline__ v16u frag2(const unsigned short* p0, const unsigned short* p1) {
  union { v8u h[2]; v16u v; } u;
  u.h[0] = *(const v8u*)p0;
  u.h[1] = *(const v8u*)p1;
  return u.v;
}

// Problem constants
#define BB_   2
#define CIN_  512
#define C1_   256
#define C2_   64
#define HH_   96
#define WW_   96
#define PP_   9216      // H*W
#define MM_   18432     // B*H*W
#define NK_   9

// ---------------- weight pre-conversion to bf16 ----------------
// w1b[c1][cin]; wdb[n][o][c] (transposed from wd[o][c][n])
__global__ __launch_bounds__(256) void prep_weights_kernel(
    const float* __restrict__ w1, const float* __restrict__ wd,
    unsigned short* __restrict__ w1b, unsigned short* __restrict__ wdb) {
  int i = blockIdx.x * 256 + threadIdx.x;           // grid covers 147456
  if (i < C1_ * CIN_) w1b[i] = f2bf(w1[i]);
  if (i < NK_ * C2_ * C1_) {
    int c = i & 255; int t = i >> 8; int o = t & 63; int n = t >> 6;
    wdb[i] = f2bf(wd[((size_t)o * C1_ + c) * NK_ + n]);
  }
}

// ---------------- BN1 stats (x layout (b,c,p)) ----------------
__global__ __launch_bounds__(256) void bn1_stats_kernel(
    const float* __restrict__ x, const float* __restrict__ g, const float* __restrict__ bt,
    float* __restrict__ scale, float* __restrict__ shift) {
  __shared__ float ssum[256], ssq[256];
  int c = blockIdx.x;
  float sum = 0.f, sq = 0.f;
  for (int b = 0; b < BB_; b++) {
    const float* xp = x + ((size_t)(b * CIN_ + c)) * PP_;
    for (int p = threadIdx.x; p < PP_; p += 256) { float v = xp[p]; sum += v; sq += v * v; }
  }
  ssum[threadIdx.x] = sum; ssq[threadIdx.x] = sq;
  __syncthreads();
  for (int off = 128; off > 0; off >>= 1) {
    if (threadIdx.x < off) { ssum[threadIdx.x] += ssum[threadIdx.x + off]; ssq[threadIdx.x] += ssq[threadIdx.x + off]; }
    __syncthreads();
  }
  if (threadIdx.x == 0) {
    float mean = ssum[0] / (float)MM_;
    float var  = ssq[0] / (float)MM_ - mean * mean;
    float sc = g[c] * rsqrtf(var + 1e-5f);
    scale[c] = sc;
    shift[c] = bt[c] - mean * sc;
  }
}

// ---------------- conv1: 1x1 GEMM with fused BN1+ReLU, WMMA bf16 ----------------
// D[c1, m] = W1[c1, cin] @ relu(bn1(x))[cin, m] + b1 ; f layout (c1, M), M = b*9216 + p
__global__ __launch_bounds__(256) void conv1_gemm_kernel(
    const float* __restrict__ x, const unsigned short* __restrict__ w1b,
    const float* __restrict__ b1,
    const float* __restrict__ scale1, const float* __restrict__ shift1, float* __restrict__ f) {
  __shared__ __align__(16) unsigned short As[64][40];   // [o][k]  row stride 80B (16B mult)
  __shared__ __align__(16) unsigned short Bs[64][40];   // [m][k]  row stride 80B
  const int tid  = threadIdx.x;
  const int wave = tid >> 5, lane = tid & 31;
  const int o_sub  = wave & 3;     // 16-row band
  const int m_half = wave >> 2;    // 32-col band (2 subtiles)
  const int tmi = blockIdx.x;      // 0..287 m-tiles
  const int c1b = blockIdx.y;      // 0..3 o-tiles
  const int bb  = tmi / 144;
  const int p0  = (tmi % 144) * 64;
  const int mbase = bb * PP_ + p0;
  const int lo = lane & 15, hi = lane >> 4;
  v8f acc0 = {}; v8f acc1 = {};
  for (int k0 = 0; k0 < CIN_; k0 += 32) {
    // stage A: 64x32 bf16, 16B chunks: 256 chunks -> 1 per thread
    {
      int o = tid >> 2, cc = (tid & 3) * 8;
      *(v8u*)&As[o][cc] = *(const v8u*)&w1b[(size_t)(c1b * 64 + o) * CIN_ + k0 + cc];
    }
    // stage B: 32k x 64m, fused bn1+relu ; coalesced global reads, [m][k] LDS layout
    for (int r = 0; r < 8; r++) {
      int e = r * 256 + tid;
      int kk = e >> 6, mm = e & 63;
      int cch = k0 + kk;
      float v = x[((size_t)(bb * CIN_ + cch)) * PP_ + p0 + mm];
      v = fmaxf(v * scale1[cch] + shift1[cch], 0.f);
      Bs[mm][kk] = f2bf(v);
    }
    __syncthreads();
    const int arow = o_sub * 16 + lo;
    const int ah = hi << 3;              // A: elems 0-7 -> k=ah.., 8-15 -> k=16+ah..
    const int bh = hi << 3;              // B: elems 0-7 -> k=bh.., 8-15 -> k=16+bh..
    const int bc0 = m_half * 32 + lo;
    v16u au  = frag2(&As[arow][ah], &As[arow][16 + ah]);
    v16u bu0 = frag2(&Bs[bc0][bh], &Bs[bc0][bh + 16]);
    v16u bu1 = frag2(&Bs[bc0 + 16][bh], &Bs[bc0 + 16][bh + 16]);
    v16bf a = __builtin_bit_cast(v16bf, au);
    acc0 = __builtin_amdgcn_wmma_f32_16x16x32_bf16(false, a, false, __builtin_bit_cast(v16bf, bu0),
                                                   (short)0, acc0, false, false);
    acc1 = __builtin_amdgcn_wmma_f32_16x16x32_bf16(false, a, false, __builtin_bit_cast(v16bf, bu1),
                                                   (short)0, acc1, false, false);
    __syncthreads();
  }
  const int ml = m_half * 32 + lo;
  #pragma unroll
  for (int r = 0; r < 8; r++) {
    int o = c1b * 64 + o_sub * 16 + r + hi * 8;
    float bias = b1[o];
    f[(size_t)o * MM_ + mbase + ml]      = acc0[r] + bias;
    f[(size_t)o * MM_ + mbase + ml + 16] = acc1[r] + bias;
  }
}

// ---------------- BN2 stats (f layout (c, M)) ----------------
__global__ __launch_bounds__(256) void bn2_stats_kernel(
    const float* __restrict__ f, const float* __restrict__ g, const float* __restrict__ bt,
    float* __restrict__ scale, float* __restrict__ shift) {
  __shared__ float ssum[256], ssq[256];
  int c = blockIdx.x;
  const float* fp = f + (size_t)c * MM_;
  float sum = 0.f, sq = 0.f;
  for (int p = threadIdx.x; p < MM_; p += 256) { float v = fp[p]; sum += v; sq += v * v; }
  ssum[threadIdx.x] = sum; ssq[threadIdx.x] = sq;
  __syncthreads();
  for (int off = 128; off > 0; off >>= 1) {
    if (threadIdx.x < off) { ssum[threadIdx.x] += ssum[threadIdx.x + off]; ssq[threadIdx.x] += ssq[threadIdx.x + off]; }
    __syncthreads();
  }
  if (threadIdx.x == 0) {
    float mean = ssum[0] / (float)MM_;
    float var  = ssq[0] / (float)MM_ - mean * mean;
    float sc = g[c] * rsqrtf(var + 1e-5f);
    scale[c] = sc;
    shift[c] = bt[c] - mean * sc;
  }
}

__global__ __launch_bounds__(256) void bn2_apply_kernel(
    float* __restrict__ f, const float* __restrict__ scale, const float* __restrict__ shift) {
  int i = blockIdx.x * 256 + threadIdx.x;   // C1_*MM_ elements
  int c = i / MM_;
  f[i] = f[i] * scale[c] + shift[c];
}

// ---------------- offset / modulation / dilation-scale 3x3 convs ----------------
__global__ __launch_bounds__(256) void offs_conv_kernel(
    const float* __restrict__ f,
    const float* __restrict__ pw, const float* __restrict__ pb,
    const float* __restrict__ mw, const float* __restrict__ mb,
    const float* __restrict__ dw, const float* __restrict__ db,
    float* __restrict__ ob) {
  __shared__ float wl[C1_ * 9];
  const int chunk = blockIdx.x;   // 0..35
  const int oc = blockIdx.y;      // 0..27
  const int bb = blockIdx.z;      // 0..1
  const float* wsrc; float bias;
  if (oc < 18)      { wsrc = pw + (size_t)oc * (C1_ * 9);        bias = pb[oc]; }
  else if (oc < 27) { wsrc = mw + (size_t)(oc - 18) * (C1_ * 9); bias = mb[oc - 18]; }
  else              { wsrc = dw;                                  bias = db[0]; }
  for (int i = threadIdx.x; i < C1_ * 9; i += 256) wl[i] = wsrc[i];
  __syncthreads();
  const int p = chunk * 256 + threadIdx.x;
  const int row = p / WW_, col = p % WW_;
  float acc = bias;
  for (int c = 0; c < C1_; c++) {
    const float* fp = f + (size_t)c * MM_ + bb * PP_;
    const float* wc = wl + c * 9;
    #pragma unroll
    for (int dr = -1; dr <= 1; dr++) {
      int rr = row + dr;
      if ((unsigned)rr >= (unsigned)HH_) continue;
      #pragma unroll
      for (int dc = -1; dc <= 1; dc++) {
        int cc = col + dc;
        if ((unsigned)cc >= (unsigned)WW_) continue;
        acc += wc[(dr + 1) * 3 + (dc + 1)] * fp[rr * WW_ + cc];
      }
    }
  }
  float v;
  if (oc < 18) v = acc;
  else if (oc < 27) v = sigmoidf_(acc);
  else v = 2.0f * sigmoidf_(acc);
  ob[(size_t)oc * MM_ + bb * PP_ + p] = v;
}

// ---------------- fused deformable sampling + final GEMM (64o x 32m tiles) ----------------
__global__ __launch_bounds__(256) void deform_gemm_kernel(
    const float* __restrict__ f, const float* __restrict__ ob,
    const unsigned short* __restrict__ wdb, float* __restrict__ out) {
  __shared__ __align__(16) unsigned short As[64][264];  // [o][c], row stride 528B (16B mult)
  __shared__ __align__(16) unsigned short Bs[32][264];  // [m][c]
  __shared__ int   sidx[32][NK_][4];
  __shared__ float sg[32][NK_][4];
  const int tid = threadIdx.x;
  const int tmi = blockIdx.x;              // 0..575
  const int bb = tmi / 288;
  const int p0 = (tmi % 288) * 32;

  // 1) sampling coordinates + bilinear weights (x modulation) for 32 pos x 9 taps
  for (int j = tid; j < 32 * NK_; j += 256) {
    int i = j / NK_, n = j % NK_;
    int p = p0 + i;
    int m = bb * PP_ + p;
    int row = p / WW_, col = p % WW_;
    float s    = ob[27 * MM_ + m];
    float offx = ob[(size_t)n * MM_ + m];
    float offy = ob[(size_t)(9 + n) * MM_ + m];
    float mval = ob[(size_t)(18 + n) * MM_ + m];
    float pnx = (float)(n / 3 - 1), pny = (float)(n % 3 - 1);
    float px = (float)(row + 1) + 6.0f * s * pnx + offx;
    float py = (float)(col + 1) + 6.0f * s * pny + offy;
    float fx = floorf(px), fy = floorf(py);
    int qxl = min(max((int)fx, 0), 97);
    int qyl = min(max((int)fy, 0), 97);
    int qxr = min(max((int)fx + 1, 0), 97);
    int qyr = min(max((int)fy + 1, 0), 97);
    float pxc = fminf(fmaxf(px, 0.f), 97.f);
    float pyc = fminf(fmaxf(py, 0.f), 97.f);
    float glt = (1.f + (float)qxl - pxc) * (1.f + (float)qyl - pyc);
    float grb = (1.f - ((float)qxr - pxc)) * (1.f - ((float)qyr - pyc));
    float glb = (1.f + (float)qxl - pxc) * (1.f - ((float)qyr - pyc));
    float grt = (1.f - ((float)qxr - pxc)) * (1.f + (float)qyl - pyc);
    sg[i][n][0] = glt * mval; sg[i][n][1] = grb * mval;
    sg[i][n][2] = glb * mval; sg[i][n][3] = grt * mval;
    #define CIDX(qx, qy) (((qx) >= 1 && (qx) <= HH_ && (qy) >= 1 && (qy) <= WW_) ? ((qx) - 1) * WW_ + ((qy) - 1) : -1)
    sidx[i][n][0] = CIDX(qxl, qyl);
    sidx[i][n][1] = CIDX(qxr, qyr);
    sidx[i][n][2] = CIDX(qxl, qyr);
    sidx[i][n][3] = CIDX(qxr, qyl);
    #undef CIDX
  }
  __syncthreads();

  const int wave = tid >> 5, lane = tid & 31;
  const int o_sub = wave & 3;      // 0..3
  const int m_sub = wave >> 2;     // 0..1
  const int lo = lane & 15, hi = lane >> 4;
  v8f acc = {};
  const int q32 = tid >> 5;        // B staging: 8 channel groups of 32
  const int ip  = tid & 31;        // position within tile
  const float* fb = f + bb * PP_;

  for (int n = 0; n < NK_; n++) {
    // stage A: 64x256 bf16 copy from pre-converted wdb[n] ; 16B chunks, 8 per thread
    {
      const unsigned short* wn = wdb + (size_t)n * (C2_ * C1_);
      #pragma unroll
      for (int r = 0; r < 8; r++) {
        int id = r * 256 + tid;            // 2048 chunks
        int o = id >> 5, cc = (id & 31) * 8;
        *(v8u*)&As[o][cc] = *(const v8u*)&wn[o * C1_ + cc];
      }
    }
    // stage B: bilinear gather + modulation -> bf16, pair-packed stores
    {
      float g0 = sg[ip][n][0], g1 = sg[ip][n][1], g2 = sg[ip][n][2], g3 = sg[ip][n][3];
      int i0 = sidx[ip][n][0], i1 = sidx[ip][n][1], i2 = sidx[ip][n][2], i3 = sidx[ip][n][3];
      for (int c = q32 * 32; c < q32 * 32 + 32; c += 2) {
        const float* fc0 = fb + (size_t)c * MM_;
        const float* fc1 = fc0 + MM_;
        float v0 = 0.f, v1 = 0.f;
        if (i0 >= 0) { v0 += g0 * fc0[i0]; v1 += g0 * fc1[i0]; }
        if (i1 >= 0) { v0 += g1 * fc0[i1]; v1 += g1 * fc1[i1]; }
        if (i2 >= 0) { v0 += g2 * fc0[i2]; v1 += g2 * fc1[i2]; }
        if (i3 >= 0) { v0 += g3 * fc0[i3]; v1 += g3 * fc1[i3]; }
        *(unsigned*)&Bs[ip][c] = pack2bf(v0, v1);
      }
    }
    __syncthreads();
    const int arow = o_sub * 16 + lo;
    const int ah = hi << 3;
    const int bh = hi << 3;
    const int bc = m_sub * 16 + lo;
    #pragma unroll
    for (int kb = 0; kb < C1_; kb += 32) {
      v16u au = frag2(&As[arow][kb + ah], &As[arow][kb + 16 + ah]);
      v16u bu = frag2(&Bs[bc][kb + bh], &Bs[bc][kb + bh + 16]);
      acc = __builtin_amdgcn_wmma_f32_16x16x32_bf16(false, __builtin_bit_cast(v16bf, au),
                                                    false, __builtin_bit_cast(v16bf, bu),
                                                    (short)0, acc, false, false);
    }
    __syncthreads();
  }
  // epilogue: D layout — VGPR r, lanes 0-15: M=r; lanes 16-31: M=r+8
  const int ml = m_sub * 16 + lo;
  #pragma unroll
  for (int r = 0; r < 8; r++) {
    int o = o_sub * 16 + r + hi * 8;
    out[((size_t)(bb * C2_ + o)) * PP_ + p0 + ml] = acc[r];
  }
}

// ---------------- host launcher ----------------
extern "C" void kernel_launch(void* const* d_in, const int* in_sizes, int n_in,
                              void* d_out, int out_size, void* d_ws, size_t ws_size,
                              hipStream_t stream) {
  (void)in_sizes; (void)n_in; (void)out_size; (void)ws_size;
  const float* x      = (const float*)d_in[0];
  const float* bn1_g  = (const float*)d_in[1];
  const float* bn1_b  = (const float*)d_in[2];
  const float* conv1w = (const float*)d_in[3];
  const float* conv1b = (const float*)d_in[4];
  const float* bn2_g  = (const float*)d_in[5];
  const float* bn2_b  = (const float*)d_in[6];
  const float* p_w    = (const float*)d_in[7];
  const float* p_b    = (const float*)d_in[8];
  const float* m_w    = (const float*)d_in[9];
  const float* m_b    = (const float*)d_in[10];
  const float* d_w    = (const float*)d_in[11];
  const float* d_b    = (const float*)d_in[12];
  const float* dconvw = (const float*)d_in[13];
  float* out = (float*)d_out;

  float* ws = (float*)d_ws;
  float* scale1 = ws;                                    // 512
  float* shift1 = ws + 512;                              // 512
  float* scale2 = ws + 1024;                             // 256
  float* shift2 = ws + 1280;                             // 256
  float* fbuf   = ws + 2048;                             // C1_*MM_ = 4,718,592 floats
  float* obuf   = fbuf + (size_t)C1_ * MM_;              // 28*MM_ = 516,096 floats
  unsigned short* w1b = (unsigned short*)(obuf + (size_t)28 * MM_);  // 131,072 bf16
  unsigned short* wdb = w1b + (size_t)C1_ * CIN_;                    // 147,456 bf16

  prep_weights_kernel<<<576, 256, 0, stream>>>(conv1w, dconvw, w1b, wdb);
  bn1_stats_kernel<<<CIN_, 256, 0, stream>>>(x, bn1_g, bn1_b, scale1, shift1);
  conv1_gemm_kernel<<<dim3(288, 4), 256, 0, stream>>>(x, w1b, conv1b, scale1, shift1, fbuf);
  bn2_stats_kernel<<<C1_, 256, 0, stream>>>(fbuf, bn2_g, bn2_b, scale2, shift2);
  bn2_apply_kernel<<<(C1_ * MM_) / 256, 256, 0, stream>>>(fbuf, scale2, shift2);
  offs_conv_kernel<<<dim3(36, 28, 2), 256, 0, stream>>>(fbuf, p_w, p_b, m_w, m_b, d_w, d_b, obuf);
  deform_gemm_kernel<<<576, 256, 0, stream>>>(fbuf, obuf, wdb, out);
}
